// FeatTransform_3693671874622
// MI455X (gfx1250) — compile-verified, matrix operand order checked by reading, not run on hardware
//
#include <hip/hip_runtime.h>

typedef __attribute__((ext_vector_type(16))) __bf16 v16bf;
typedef __attribute__((ext_vector_type(8)))  __bf16 v8bf;
typedef __attribute__((ext_vector_type(8)))  float  v8f;

#define FDIM 128
#define NEG_SLOPE 0.2f
// LDS row stride for staged weights: 128 + 8 bf16 -> 272 B = 68 DWORDs.
// 68 mod 64 = 4 banks/row => 16 consecutive columns map to distinct bank
// groups for ds_load_b128 (no 16-way conflicts).
#define LDSTRIDE 136
#define LDS_BYTES (2 * FDIM * LDSTRIDE * 2)  // W1T + W2T staged, 69632 B

// ---------------------------------------------------------------------------
// Kernel 1: transpose W1/W2 (128x128 f32) into bf16 W^T so that columns of W
// (the WMMA B-operand K-runs) are contiguous in memory.
// ---------------------------------------------------------------------------
__global__ __launch_bounds__(256) void gnn_prep_weights(
    const float* __restrict__ W1, const float* __restrict__ W2,
    __bf16* __restrict__ W1T, __bf16* __restrict__ W2T) {
  const int idx = blockIdx.x * blockDim.x + threadIdx.x;
  if (idx >= FDIM * FDIM) return;
  const int r = idx >> 7;            // k
  const int c = idx & (FDIM - 1);    // n
  W1T[c * FDIM + r] = (__bf16)W1[idx];
  W2T[c * FDIM + r] = (__bf16)W2[idx];
}

// ---------------------------------------------------------------------------
// Kernel 2: SpMM scatter. One wave32 per edge: lane l handles features
// 4l..4l+3 -> one coalesced global_load_b128 of the source row (L2-resident)
// and four global_atomic_add_f32 into the destination row (resolved in L2).
// Edge arrays are streamed once -> non-temporal loads keep them from
// evicting the feat/agg working set out of the 192 MB L2.
// ---------------------------------------------------------------------------
__global__ __launch_bounds__(256) void gnn_spmm_scatter(
    const int* __restrict__ esrc, const int* __restrict__ edst,
    const float* __restrict__ evals, const float* __restrict__ feat,
    float* __restrict__ agg, int E) {
  const int lane = threadIdx.x & 31;
  const int e = blockIdx.x * (blockDim.x >> 5) + (threadIdx.x >> 5);
  if (e >= E) return;
  const int   s = __builtin_nontemporal_load(esrc + e);
  const int   d = __builtin_nontemporal_load(edst + e);
  const float v = __builtin_nontemporal_load(evals + e);
  const float4 m = *(const float4*)(feat + (size_t)s * FDIM + lane * 4);
  float* ap = agg + (size_t)d * FDIM + lane * 4;
  atomicAdd(ap + 0, v * m.x);
  atomicAdd(ap + 1, v * m.y);
  atomicAdd(ap + 2, v * m.z);
  atomicAdd(ap + 3, v * m.w);
}

// ---------------------------------------------------------------------------
// Kernel 3: fused dual-GEMM + bias + leaky_relu + add.
// One wave per 16-row tile (8 waves / block). Weights are staged once per
// block into LDS (bank-padded) and B operands come from ds_load_b128; A
// operands are built in-register in the CDNA5 16-bit A layout:
//   lanes 0-15  hold M=lane,    K = {0..7, 16..23}
//   lanes 16-31 hold M=lane-16, K = {8..15, 24..31}   (per 32-wide K tile)
// ---------------------------------------------------------------------------
__device__ __forceinline__ v16bf loadB_lds(const __bf16* p) {
  v8bf lo = *(const v8bf*)p;
  v8bf hi = *(const v8bf*)(p + 16);
  v16bf r;
#pragma unroll
  for (int i = 0; i < 8; ++i) { r[i] = lo[i]; r[i + 8] = hi[i]; }
  return r;
}

__global__ __launch_bounds__(256) void gnn_fused_gemm(
    const float* __restrict__ agg, const float* __restrict__ feat,
    const __bf16* __restrict__ W1T, const __bf16* __restrict__ W2T,
    const float* __restrict__ b1, const float* __restrict__ b2,
    float* __restrict__ out, int ntiles, int N) {
  extern __shared__ __bf16 smem[];
  __bf16* sW1 = smem;                       // [FDIM][LDSTRIDE]
  __bf16* sW2 = smem + FDIM * LDSTRIDE;     // [FDIM][LDSTRIDE]

  // ---- Stage W1T/W2T into LDS: 16B global loads -> 16B ds stores. ----
  // 128 rows x 16 chunks of 8 bf16 = 2048 chunks per matrix; 256 threads.
  {
    const int tid = threadIdx.x;
#pragma unroll
    for (int it = 0; it < 8; ++it) {
      const int ch  = it * 256 + tid;
      const int row = ch >> 4;
      const int off = (ch & 15) * 8;
      *(v8bf*)(sW1 + row * LDSTRIDE + off) = *(const v8bf*)(W1T + row * FDIM + off);
      *(v8bf*)(sW2 + row * LDSTRIDE + off) = *(const v8bf*)(W2T + row * FDIM + off);
    }
  }
  __syncthreads();

  const int lane = threadIdx.x & 31;
  const int tile = blockIdx.x * (blockDim.x >> 5) + (threadIdx.x >> 5);
  if (tile >= ntiles) return;
  const int half = lane >> 4;
  const int l16  = lane & 15;
  const int kb   = half * 8;

  int row = tile * 16 + l16;
  if (row >= N) row = N - 1;  // clamp: duplicate loads, stores guarded below

  const float* arow = agg  + (size_t)row * FDIM;
  const float* frow = feat + (size_t)row * FDIM;

  // Build bf16 A operands for both GEMMs: Ag = agg, Ax = agg * feat.
  v16bf Ag[4], Ax[4];
#pragma unroll
  for (int kt = 0; kt < 4; ++kt) {
    const int k0 = kt * 32 + kb;
    float4 a0 = *(const float4*)(arow + k0);
    float4 a1 = *(const float4*)(arow + k0 + 4);
    float4 a2 = *(const float4*)(arow + k0 + 16);
    float4 a3 = *(const float4*)(arow + k0 + 20);
    float4 f0 = *(const float4*)(frow + k0);
    float4 f1 = *(const float4*)(frow + k0 + 4);
    float4 f2 = *(const float4*)(frow + k0 + 16);
    float4 f3 = *(const float4*)(frow + k0 + 20);

    Ag[kt][0]  = (__bf16)a0.x; Ag[kt][1]  = (__bf16)a0.y;
    Ag[kt][2]  = (__bf16)a0.z; Ag[kt][3]  = (__bf16)a0.w;
    Ag[kt][4]  = (__bf16)a1.x; Ag[kt][5]  = (__bf16)a1.y;
    Ag[kt][6]  = (__bf16)a1.z; Ag[kt][7]  = (__bf16)a1.w;
    Ag[kt][8]  = (__bf16)a2.x; Ag[kt][9]  = (__bf16)a2.y;
    Ag[kt][10] = (__bf16)a2.z; Ag[kt][11] = (__bf16)a2.w;
    Ag[kt][12] = (__bf16)a3.x; Ag[kt][13] = (__bf16)a3.y;
    Ag[kt][14] = (__bf16)a3.z; Ag[kt][15] = (__bf16)a3.w;

    Ax[kt][0]  = (__bf16)(a0.x * f0.x); Ax[kt][1]  = (__bf16)(a0.y * f0.y);
    Ax[kt][2]  = (__bf16)(a0.z * f0.z); Ax[kt][3]  = (__bf16)(a0.w * f0.w);
    Ax[kt][4]  = (__bf16)(a1.x * f1.x); Ax[kt][5]  = (__bf16)(a1.y * f1.y);
    Ax[kt][6]  = (__bf16)(a1.z * f1.z); Ax[kt][7]  = (__bf16)(a1.w * f1.w);
    Ax[kt][8]  = (__bf16)(a2.x * f2.x); Ax[kt][9]  = (__bf16)(a2.y * f2.y);
    Ax[kt][10] = (__bf16)(a2.z * f2.z); Ax[kt][11] = (__bf16)(a2.w * f2.w);
    Ax[kt][12] = (__bf16)(a3.x * f3.x); Ax[kt][13] = (__bf16)(a3.y * f3.y);
    Ax[kt][14] = (__bf16)(a3.z * f3.z); Ax[kt][15] = (__bf16)(a3.w * f3.w);
  }

  const int rbase = tile * 16 + half * 8;
#pragma unroll
  for (int nt = 0; nt < 8; ++nt) {
    const int col = nt * 16 + l16;
    const __bf16* w1 = sW1 + col * LDSTRIDE + kb;
    const __bf16* w2 = sW2 + col * LDSTRIDE + kb;
    v8f acc1 = {};
    v8f acc2 = {};
#pragma unroll
    for (int kt = 0; kt < 4; ++kt) {
      acc1 = __builtin_amdgcn_wmma_f32_16x16x32_bf16(
          false, Ag[kt], false, loadB_lds(w1 + kt * 32), (short)0, acc1, false, false);
      acc2 = __builtin_amdgcn_wmma_f32_16x16x32_bf16(
          false, Ax[kt], false, loadB_lds(w2 + kt * 32), (short)0, acc2, false, false);
    }
    const float bb1 = b1[col];
    const float bb2 = b2[col];
#pragma unroll
    for (int v = 0; v < 8; ++v) {
      const int r = rbase + v;  // f32 C/D layout: VGPR v -> M = v + 8*half
      if (r < N) {
        float x1 = acc1[v] + bb1; x1 = (x1 >= 0.0f) ? x1 : NEG_SLOPE * x1;
        float x2 = acc2[v] + bb2; x2 = (x2 >= 0.0f) ? x2 : NEG_SLOPE * x2;
        __builtin_nontemporal_store(x1 + x2, out + (size_t)r * FDIM + col);
      }
    }
  }
}

// ---------------------------------------------------------------------------
extern "C" void kernel_launch(void* const* d_in, const int* in_sizes, int n_in,
                              void* d_out, int out_size, void* d_ws, size_t ws_size,
                              hipStream_t stream) {
  const int*   esrc  = (const int*)d_in[0];
  const int*   edst  = (const int*)d_in[1];
  const float* evals = (const float*)d_in[2];
  const float* feat  = (const float*)d_in[3];
  const float* W1    = (const float*)d_in[4];
  const float* b1    = (const float*)d_in[5];
  const float* W2    = (const float*)d_in[6];
  const float* b2    = (const float*)d_in[7];

  const int E = in_sizes[0];
  const int N = in_sizes[3] / FDIM;

  // Workspace: agg [N*128 f32] | W1T [128*128 bf16] | W2T [128*128 bf16]
  float* agg = (float*)d_ws;
  const size_t aggBytes = (size_t)N * FDIM * sizeof(float);  // 256B-aligned
  __bf16* W1T = (__bf16*)((char*)d_ws + aggBytes);
  __bf16* W2T = W1T + FDIM * FDIM;

  hipMemsetAsync(agg, 0, aggBytes, stream);

  gnn_prep_weights<<<(FDIM * FDIM + 255) / 256, 256, 0, stream>>>(W1, W2, W1T, W2T);

  const int wavesPerBlock = 8;  // 256 threads
  const int eblocks = (E + wavesPerBlock - 1) / wavesPerBlock;
  gnn_spmm_scatter<<<eblocks, 256, 0, stream>>>(esrc, edst, evals, feat, agg, E);

  const int ntiles = (N + 15) / 16;
  const int gblocks = (ntiles + wavesPerBlock - 1) / wavesPerBlock;
  gnn_fused_gemm<<<gblocks, 256, LDS_BYTES, stream>>>(agg, feat, W1T, W2T, b1, b2,
                                                      (float*)d_out, ntiles, N);
}
